// GraphConvClf_79345225826538
// MI455X (gfx1250) — matrix operand order repaired and checked by reading, not compiled, not run on hardware
//
#include <hip/hip_runtime.h>
#include <hip/hip_bf16.h>

typedef __attribute__((ext_vector_type(16))) _Float16 v16h;
typedef __attribute__((ext_vector_type(8)))  float    v8f;

#define NB    32
#define VPG   8192
#define NVERT (NB*VPG)        // 262144
#define EPG   24576
#define NEDGE (NB*EPG)        // 786432
#define NEGV  (-1e30f)

__device__ __forceinline__ unsigned okey(float f) {
  unsigned u = __float_as_uint(f);
  return (u & 0x80000000u) ? ~u : (u | 0x80000000u);   // ascending uint == ascending float
}

// ---------------- utility ----------------
__global__ void k_set_mask(int* vmask, int n) {
  int i = blockIdx.x * blockDim.x + threadIdx.x;
  if (i < n) vmask[i] = 1;
}
__global__ void k_zero(float* p, int n) {
  int i = blockIdx.x * blockDim.x + threadIdx.x;
  if (i < n) p[i] = 0.0f;
}

// ---------------- score matvec: score = x@w0 + b0 (or NEG), yv = x@w1 + b1 ----------------
__global__ void k_score(const float* __restrict__ X, int lda, int D,
                        const float* __restrict__ w0, const float* __restrict__ b0,
                        const float* __restrict__ w1, const float* __restrict__ b1,
                        const int* __restrict__ vmask,
                        float* __restrict__ score, float* __restrict__ yv, int n) {
  int i = blockIdx.x * blockDim.x + threadIdx.x;
  if (i >= n) return;
  if (!vmask[i]) { score[i] = NEGV; yv[i] = 0.0f; return; }
  const float* xr = X + (size_t)i * lda;
  float s0 = b0[0], s1 = b1[0];
  for (int k = 0; k < D; ++k) { float v = xr[k]; s0 += v * w0[k]; s1 += v * w1[k]; }
  score[i] = s0; yv[i] = s1;
}

// ---------------- scalar edge scatter for scores ----------------
__global__ void k_scatter_s(const int2* __restrict__ edges, const int* __restrict__ vmask,
                            const float* __restrict__ yv, float* __restrict__ score, int ne) {
  int e = blockIdx.x * blockDim.x + threadIdx.x;
  if (e >= ne) return;
  int2 ed = edges[e];
  if (vmask[ed.x] && vmask[ed.y]) {
    atomicAdd(&score[ed.x], yv[ed.y]);
    atomicAdd(&score[ed.y], yv[ed.x]);
  }
}

// ---------------- per-graph top-k via 4-pass radix select (LDS histogram) ----------------
__global__ void k_topk(const float* __restrict__ score, int* __restrict__ vmask, int kkeep) {
  __shared__ unsigned hist[256];
  __shared__ unsigned s_pref;
  __shared__ int s_rem;
  __shared__ unsigned s_eq;
  int tid = threadIdx.x;
  int base = blockIdx.x * VPG;
  if (tid == 0) { s_pref = 0u; s_rem = kkeep; s_eq = 0u; }
  __syncthreads();
  for (int pass = 0; pass < 4; ++pass) {
    int shift = 24 - 8 * pass;
    hist[tid] = 0u;
    __syncthreads();
    unsigned pref = s_pref;
    for (int v = tid; v < VPG; v += 256) {
      unsigned key = okey(score[base + v]);
      bool match = (pass == 0) || ((key >> (shift + 8)) == pref);
      if (match) atomicAdd(&hist[(key >> shift) & 0xFFu], 1u);
    }
    __syncthreads();
    if (tid == 0) {
      int rem = s_rem; unsigned cum = 0; unsigned chosen = 0;
      for (int b = 255; b >= 0; --b) {
        unsigned h = hist[b];
        if (cum + h >= (unsigned)rem) { chosen = (unsigned)b; rem -= (int)cum; break; }
        cum += h;
      }
      s_pref = (pref << 8) | chosen;
      s_rem = rem;
    }
    __syncthreads();
  }
  unsigned T = s_pref; int need = s_rem;
  for (int v = tid; v < VPG; v += 256) {
    int n = base + v;
    int kp = 0;
    if (vmask[n]) {
      unsigned key = okey(score[n]);
      if (key > T) kp = 1;
      else if (key == T) { unsigned o = atomicAdd(&s_eq, 1u); if (o < (unsigned)need) kp = 1; }
    }
    vmask[n] = kp;
  }
}

// ---------------- tscore = kept ? tanh(score) : 0 (hoisted transcendental) ----------------
__global__ void k_tanh(const float* __restrict__ score, const int* __restrict__ vmask,
                       float* __restrict__ tscore, int n) {
  int i = blockIdx.x * blockDim.x + threadIdx.x;
  if (i >= n) return;
  tscore[i] = vmask[i] ? tanhf(score[i]) : 0.0f;
}

// ---------------- x = kept ? relu(x * tscore) : 0 ----------------
__global__ void k_xupdate(const float* __restrict__ Xin, const float* __restrict__ tscore,
                          const int* __restrict__ vmask, float* __restrict__ Xout,
                          int D, long long total) {
  long long i = (long long)blockIdx.x * blockDim.x + threadIdx.x;
  if (i >= total) return;
  int n = (int)(i / D);
  float v = 0.0f;
  if (vmask[n]) v = fmaxf(Xin[i] * tscore[n], 0.0f);
  Xout[i] = v;
}

// ---------------- WMMA GEMM: Out(N,128) = X(N,K) @ W(128,K)^T + bias ----------------
// One 256-thread block = 8 waves; wave w computes rows [blk*128+w*16, +16) x all 128 cols.
// A/B fragments use contiguous k order: lane half h holds k = 16h..16h+15 in both A and B.
__global__ void __launch_bounds__(256) k_gemm(const float* __restrict__ X, int K,
                                              const float* __restrict__ W,
                                              const float* __restrict__ bias,
                                              float* __restrict__ Out) {
  __shared__ __align__(32) _Float16 Wh[128 * 128];
  int tid = threadIdx.x;
  for (int idx = tid; idx < 128 * K; idx += 256) Wh[idx] = (_Float16)W[idx];
  __syncthreads();

  int lane  = tid & 31;
  int wave  = tid >> 5;
  int nn    = lane & 15;        // A row within tile / B,C,D column
  int khalf = lane >> 4;        // which K half-group this lane holds
  int kbase = khalf * 16;       // A: lanes>=16 hold K+16..
  int koff  = khalf * 16;       // B: lanes>=16 hold K+16..
  int rowbase = blockIdx.x * 128 + wave * 16;
  const float* xrow = X + (size_t)(rowbase + nn) * K;

  v8f acc[8];
#pragma unroll
  for (int t = 0; t < 8; ++t) {
    float bv = bias[t * 16 + nn];
#pragma unroll
    for (int r = 0; r < 8; ++r) acc[t][r] = bv;
  }

  for (int k0 = 0; k0 < K; k0 += 32) {
    __builtin_prefetch(xrow + k0 + 32, 0, 3);   // global_prefetch_b8, near scope
    float buf[16];
    *(float4*)&buf[0]  = *(const float4*)(xrow + k0 + kbase);
    *(float4*)&buf[4]  = *(const float4*)(xrow + k0 + kbase + 4);
    *(float4*)&buf[8]  = *(const float4*)(xrow + k0 + kbase + 8);
    *(float4*)&buf[12] = *(const float4*)(xrow + k0 + kbase + 12);
    v16h a;
#pragma unroll
    for (int j = 0; j < 16; ++j) a[j] = (_Float16)buf[j];
#pragma unroll
    for (int t = 0; t < 8; ++t) {
      v16h bf = *(const v16h*)&Wh[(size_t)(t * 16 + nn) * K + k0 + koff];
      acc[t] = __builtin_amdgcn_wmma_f32_16x16x32_f16(false, a, false, bf,
                                                      (short)0, acc[t], false, false);
    }
  }

  float* orow = Out + (size_t)(rowbase + khalf * 8) * 128 + nn;
#pragma unroll
  for (int t = 0; t < 8; ++t) {
#pragma unroll
    for (int r = 0; r < 8; ++r)
      orow[(size_t)r * 128 + t * 16] = acc[t][r];
  }
}

// ---------------- 128-wide edge scatter: O[e0]+=Y[e1], O[e1]+=Y[e0] (active edges) ----------------
__global__ void k_scatter_v(const int2* __restrict__ edges, const int* __restrict__ vmask,
                            const float* __restrict__ Y, float* __restrict__ O, long long total) {
  long long i = (long long)blockIdx.x * blockDim.x + threadIdx.x;
  if (i >= total) return;
  int e = (int)(i >> 5);
  int c = ((int)i & 31) * 4;
  int2 ed = edges[e];
  if (!(vmask[ed.x] && vmask[ed.y])) return;
  float4 yb = *(const float4*)&Y[(size_t)ed.y * 128 + c];
  float4 ya = *(const float4*)&Y[(size_t)ed.x * 128 + c];
  float* oa = &O[(size_t)ed.x * 128 + c];
  float* ob = &O[(size_t)ed.y * 128 + c];
  atomicAdd(oa + 0, yb.x); atomicAdd(oa + 1, yb.y); atomicAdd(oa + 2, yb.z); atomicAdd(oa + 3, yb.w);
  atomicAdd(ob + 0, ya.x); atomicAdd(ob + 1, ya.y); atomicAdd(ob + 2, ya.z); atomicAdd(ob + 3, ya.w);
}

// ---------------- X = vmask ? relu(O) : 0 ----------------
__global__ void k_relu_mask(const float* __restrict__ O, const int* __restrict__ vmask,
                            float* __restrict__ X, long long total) {
  long long i = (long long)blockIdx.x * blockDim.x + threadIdx.x;
  if (i >= total) return;
  int n = (int)(i >> 7);
  X[i] = vmask[n] ? fmaxf(O[i], 0.0f) : 0.0f;
}

// ---------------- per-feature sum / sumsq ----------------
__global__ void k_stats(const float* __restrict__ X, float* __restrict__ stats, int n) {
  int f = threadIdx.x;   // blockDim.x == 128
  float s = 0.0f, s2 = 0.0f;
  for (int r = blockIdx.x; r < n; r += gridDim.x) {
    float v = X[(size_t)r * 128 + f];
    s += v; s2 += v * v;
  }
  atomicAdd(&stats[f], s);
  atomicAdd(&stats[128 + f], s2);
}

// ---------------- batchnorm + relu + mask (in place) ----------------
__global__ void k_bn(float* __restrict__ X, const float* __restrict__ stats, float invcnt,
                     const float* __restrict__ g, const float* __restrict__ b,
                     const int* __restrict__ vmask, long long total) {
  long long i = (long long)blockIdx.x * blockDim.x + threadIdx.x;
  if (i >= total) return;
  int f = (int)(i & 127);
  int n = (int)(i >> 7);
  float mean = stats[f] * invcnt;
  float var  = stats[128 + f] * invcnt - mean * mean;
  float val  = (X[i] - mean) * rsqrtf(var + 1e-5f) * g[f] + b[f];
  X[i] = vmask[n] ? fmaxf(val, 0.0f) : 0.0f;
}

// ---------------- per-graph avg/max pool -> h(B,256) ----------------
__global__ void k_pool(const float* __restrict__ X, const int* __restrict__ vmask,
                       float* __restrict__ h, float invk) {
  __shared__ float psum[128];
  __shared__ int   pmax[128];
  int tid = threadIdx.x;
  if (tid < 128) { psum[tid] = 0.0f; pmax[tid] = 0; }
  __syncthreads();
  int f = tid & 127, half = tid >> 7;
  int base = blockIdx.x * VPG;
  float s = 0.0f, mx = 0.0f;   // active x >= 0 after relu, so 0 is a safe max identity
  for (int v = half; v < VPG; v += 2) {
    int n = base + v;
    if (vmask[n]) {
      float val = X[(size_t)n * 128 + f];
      s += val; mx = fmaxf(mx, val);
    }
  }
  atomicAdd(&psum[f], s);
  atomicMax(&pmax[f], __float_as_int(mx));   // valid for non-negative floats
  __syncthreads();
  if (tid < 128) {
    h[blockIdx.x * 256 + tid]       = psum[tid] * invk;
    h[blockIdx.x * 256 + 128 + tid] = __int_as_float(pmax[tid]);
  }
}

// ---------------- head: h1 = relu(h @ fc1^T + b1), out = h1 @ fc2^T + b2 ----------------
__global__ void k_head1(const float* __restrict__ h, const float* __restrict__ w,
                        const float* __restrict__ b, float* __restrict__ h1) {
  __shared__ float row[256];
  int r = blockIdx.x, c = threadIdx.x;
  row[c] = h[r * 256 + c];
  __syncthreads();
  const float* wr = w + (size_t)c * 256;
  float s = b[c];
  for (int k = 0; k < 256; ++k) s += row[k] * wr[k];
  h1[r * 256 + c] = fmaxf(s, 0.0f);
}
__global__ void k_head2(const float* __restrict__ h1, const float* __restrict__ w,
                        const float* __restrict__ b, float* __restrict__ out) {
  int i = blockIdx.x * blockDim.x + threadIdx.x;
  if (i >= 320) return;
  int r = i / 10, c = i % 10;
  const float* hr = h1 + (size_t)r * 256;
  const float* wr = w + (size_t)c * 256;
  float s = b[c];
  for (int k = 0; k < 256; ++k) s += hr[k] * wr[k];
  out[i] = s;
}

extern "C" void kernel_launch(void* const* d_in, const int* in_sizes, int n_in,
                              void* d_out, int out_size, void* d_ws, size_t ws_size,
                              hipStream_t stream) {
  (void)in_sizes; (void)n_in; (void)out_size; (void)ws_size;
  const float* verts = (const float*)d_in[0];
  const int2*  edges = (const int2*)d_in[1];
  auto P = [&](int j) -> const float* { return (const float*)d_in[4 + j]; };

  // workspace layout (floats)
  float* X     = (float*)d_ws;                    // N*128
  float* O     = X + (size_t)NVERT * 128;         // N*128
  float* Y     = O + (size_t)NVERT * 128;         // N*128
  float* score = Y + (size_t)NVERT * 128;         // N
  float* yv    = score + NVERT;                   // N (reused as tanh(score) after topk)
  float* h     = yv + NVERT;                      // B*256
  float* h1    = h + NB * 256;                    // B*256
  float* stats = h1 + NB * 256;                   // 256
  int*   vmask = (int*)(stats + 256);             // N ints

  k_set_mask<<<NVERT / 256, 256, 0, stream>>>(vmask, NVERT);

  const int kvals[3] = {4096, 2048, 1024};
  const float* Xin = verts;
  int Din = 64;

  for (int i = 0; i < 3; ++i) {
    int pb = i * 10;
    const float *pw0 = P(pb + 0), *pb0 = P(pb + 1), *pw1 = P(pb + 2), *pb1 = P(pb + 3);
    const float *gw0 = P(pb + 4), *gb0 = P(pb + 5), *gw1 = P(pb + 6), *gb1 = P(pb + 7);
    const float *bng = P(pb + 8), *bnb = P(pb + 9);

    k_score<<<NVERT / 256, 256, 0, stream>>>(Xin, Din, Din, pw0, pb0, pw1, pb1,
                                             vmask, score, yv, NVERT);
    k_scatter_s<<<NEDGE / 256, 256, 0, stream>>>(edges, vmask, yv, score, NEDGE);
    k_topk<<<NB, 256, 0, stream>>>(score, vmask, kvals[i]);

    // hoist transcendental: yv now holds tanh(score) for kept vertices
    k_tanh<<<NVERT / 256, 256, 0, stream>>>(score, vmask, yv, NVERT);

    long long totx = (long long)NVERT * Din;
    k_xupdate<<<(unsigned)((totx + 255) / 256), 256, 0, stream>>>(Xin, yv, vmask, X, Din, totx);

    k_gemm<<<NVERT / 128, 256, 0, stream>>>(X, Din, gw0, gb0, O);
    k_gemm<<<NVERT / 128, 256, 0, stream>>>(X, Din, gw1, gb1, Y);

    long long tots = (long long)NEDGE * 32;
    k_scatter_v<<<(unsigned)((tots + 255) / 256), 256, 0, stream>>>(edges, vmask, Y, O, tots);

    long long totf = (long long)NVERT * 128;
    k_relu_mask<<<(unsigned)(totf / 256), 256, 0, stream>>>(O, vmask, X, totf);

    k_zero<<<1, 256, 0, stream>>>(stats, 256);
    k_stats<<<1024, 128, 0, stream>>>(X, stats, NVERT);
    float invcnt = 1.0f / (float)(NB * kvals[i]);
    k_bn<<<(unsigned)(totf / 256), 256, 0, stream>>>(X, stats, invcnt, bng, bnb, vmask, totf);

    Xin = X; Din = 128;
  }

  k_pool<<<NB, 256, 0, stream>>>(X, vmask, h, 1.0f / 1024.0f);
  k_head1<<<NB, 256, 0, stream>>>(h, P(34), P(35), h1);
  k_head2<<<2, 256, 0, stream>>>(h1, P(36), P(37), (float*)d_out);
}
